// SimVQ_31267361915592
// MI455X (gfx1250) — compile-verified, matrix operand order checked by reading, not run on hardware
//
#include <hip/hip_runtime.h>

#define N_E   8192
#define EDIM  256
#define CDIM  256
#define NZ    16384   // B*H*W = 16*32*32
#define LP    264     // LDS pitch in ushorts (512B row + 16B pad)
#define TROWS 32      // codebook rows per TDM tile
#define NT    (N_E / TROWS)

typedef __attribute__((ext_vector_type(16))) __bf16 bf16x16;
typedef __attribute__((ext_vector_type(8)))  float  f32x8;
typedef __attribute__((ext_vector_type(4)))  unsigned int u32x4;
typedef __attribute__((ext_vector_type(8)))  int i32x8;
typedef __attribute__((ext_vector_type(4)))  int i32x4;

union Frag { bf16x16 v; unsigned u[8]; };

__device__ __forceinline__ unsigned short f2bf(float f) {
    unsigned x = __float_as_uint(f);
    unsigned r = ((x >> 16) & 1u) + 0x7FFFu;   // round-to-nearest-even
    return (unsigned short)((x + r) >> 16);
}

// K byte-pair base for VGPR v of a 16-bit WMMA operand, lane-half kg (0/1).
__device__ __forceinline__ int kmap(int v, int kg) {
    return ((v < 4) ? (v * 2) : (16 + (v - 4) * 2)) + kg * 8;
}

__device__ __forceinline__ float swap16_f(float x) {
    return __int_as_float(__builtin_amdgcn_ds_swizzle(__float_as_int(x), 0x401F)); // SWAPX16
}
__device__ __forceinline__ int swap16_i(int x) {
    return __builtin_amdgcn_ds_swizzle(x, 0x401F);
}

// TDM: DMA one 32x256 bf16 codebook tile into LDS with 16B/row padding (pitch 528B).
__device__ __forceinline__ void tdm_issue(const unsigned short* cbb, unsigned ldsoff, int t) {
    unsigned long long ga = (unsigned long long)(size_t)cbb +
                            (unsigned long long)t * (TROWS * CDIM * 2);
    u32x4 g0;
    g0.x = 1u;                                   // count=1, user mode
    g0.y = ldsoff;                               // lds_addr (bytes)
    g0.z = (unsigned)(ga & 0xFFFFFFFFu);         // global_addr[31:0]
    g0.w = (unsigned)((ga >> 32) & 0x01FFFFFFu)  // global_addr[56:32]
         | 0x80000000u;                          // type=2 (bits 127:126)
    // data_size=1(2B) | pad_enable | pad_interval=6(128 DW) | pad_amount=3(4 DW)
    i32x8 g1 = { 0x07910000,
                 0x01000000,                     // tensor_dim0 = 256 (bits 79:48)
                 0x20000000,                     // tensor_dim1 = 8192 (bits 111:80)
                 0x01000000,                     // tile_dim0 = 256 (bits 127:112)
                 TROWS,                          // tile_dim1 = 32
                 CDIM,                           // tensor_dim0_stride = 256
                 0, 0 };
    i32x4 gz4 = { 0, 0, 0, 0 };
    i32x8 gz8 = { 0, 0, 0, 0, 0, 0, 0, 0 };
    __builtin_amdgcn_tensor_load_to_lds(g0, g1, gz4, gz4, gz8, 0);
}

// ---------------- z: BCHW -> [N,C] fp32 + bf16 ----------------
__global__ void k_prep_z(const float* __restrict__ z, float* __restrict__ zf,
                         unsigned short* __restrict__ zb) {
    int i = blockIdx.x;            // row 0..16383
    int c = threadIdx.x;           // 0..255
    int b = i >> 10, hw = i & 1023;
    float v = z[((size_t)(b * CDIM + c) << 10) + hw];
    zf[(size_t)i * CDIM + c] = v;
    zb[(size_t)i * CDIM + c] = f2bf(v);
}

__global__ void k_cvt_bf16(const float* __restrict__ in, unsigned short* __restrict__ out, int n) {
    int i = blockIdx.x * blockDim.x + threadIdx.x;
    if (i < n) out[i] = f2bf(in[i]);
}

// ---------------- codebook = F @ W^T via WMMA bf16 ----------------
__global__ void __launch_bounds__(256) k_transform(const unsigned short* __restrict__ Fb,
                                                   const unsigned short* __restrict__ Wb,
                                                   float* __restrict__ cb) {
    int tid = threadIdx.x, lane = tid & 31, wave = tid >> 5;
    int kg = lane >> 4, ln = lane & 15;
    int rowbase = blockIdx.x * 128 + wave * 16;   // 64 blocks, 8 waves x 16 rows

    Frag A[8];
    #pragma unroll
    for (int kc = 0; kc < 8; ++kc)
        #pragma unroll
        for (int v = 0; v < 8; ++v)
            A[kc].u[v] = *(const unsigned*)(Fb + (size_t)(rowbase + ln) * EDIM + kc * 32 + kmap(v, kg));

    for (int cs = 0; cs < 16; ++cs) {
        int cbase = cs * 16;
        f32x8 acc = {0.f, 0.f, 0.f, 0.f, 0.f, 0.f, 0.f, 0.f};
        #pragma unroll
        for (int kc = 0; kc < 8; ++kc) {
            Frag Bf;
            #pragma unroll
            for (int v = 0; v < 8; ++v)
                Bf.u[v] = *(const unsigned*)(Wb + (size_t)(cbase + ln) * EDIM + kc * 32 + kmap(v, kg));
            acc = __builtin_amdgcn_wmma_f32_16x16x32_bf16(false, A[kc].v, false, Bf.v,
                                                          (short)0, acc, false, false);
        }
        #pragma unroll
        for (int v = 0; v < 8; ++v)
            cb[(size_t)(rowbase + v + 8 * kg) * CDIM + cbase + ln] = acc[v];
    }
}

// ---------------- codebook norms + bf16 copy ----------------
__global__ void k_cbnorm(const float* __restrict__ cb, unsigned short* __restrict__ cbb,
                         float* __restrict__ nrm) {
    __shared__ float s[256];
    int r = blockIdx.x, c = threadIdx.x;
    float v = cb[(size_t)r * CDIM + c];
    cbb[(size_t)r * CDIM + c] = f2bf(v);
    s[c] = v * v;
    __syncthreads();
    for (int o = 128; o > 0; o >>= 1) { if (c < o) s[c] += s[c + o]; __syncthreads(); }
    if (c == 0) nrm[r] = s[0];
}

// ---------------- fused distance GEMM + argmin (TDM double-buffered) ----------------
// score(n, m) = ||cb_m||^2 - 2 * dot(cb_m, z_n)   (||z||^2 constant under argmin)
__global__ void __launch_bounds__(256) k_dist(const unsigned short* __restrict__ zb,
                                              const unsigned short* __restrict__ cbb,
                                              const float* __restrict__ cbn,
                                              int* __restrict__ idx_out) {
    __shared__ __align__(16) unsigned short cbT[2][TROWS * LP];
    __shared__ float nrm[2][TROWS];
    __shared__ float rv[2][4][16];
    __shared__ int   ri[2][4][16];

    int tid = threadIdx.x, lane = tid & 31, wave = tid >> 5;
    int kg = lane >> 4, ln = lane & 15;
    int nsub = wave & 3, mgrp = wave >> 2;     // 4 n-subtiles x 2 m-subtiles = 8 waves
    int nbase = blockIdx.x * 64 + nsub * 16;
    int moff = mgrp * 16;

    // z fragments stay resident in VGPRs for the whole codebook sweep
    Frag Bf[8];
    #pragma unroll
    for (int kc = 0; kc < 8; ++kc)
        #pragma unroll
        for (int v = 0; v < 8; ++v)
            Bf[kc].u[v] = *(const unsigned*)(zb + (size_t)(nbase + ln) * CDIM + kc * 32 + kmap(v, kg));

    unsigned lds0 = (unsigned)(size_t)&cbT[0][0];
    unsigned lds1 = (unsigned)(size_t)&cbT[1][0];

    float bestv = 3.4e38f;
    int   besti = 0;

    if (wave == 0) {
        tdm_issue(cbb, lds0, 0);               // prime the pipeline
        nrm[0][lane] = cbn[lane];              // 32 norms for tile 0
    }

    for (int t = 0; t < NT; ++t) {
        int buf = t & 1;
        if (wave == 0) {
            if (t + 1 < NT) {
                tdm_issue(cbb, buf ? lds0 : lds1, t + 1);
                nrm[buf ^ 1][lane] = cbn[(t + 1) * TROWS + lane];
                __builtin_amdgcn_s_wait_tensorcnt(1);   // tile t complete (in-order)
            } else {
                __builtin_amdgcn_s_wait_tensorcnt(0);
            }
        }
        __syncthreads();                        // tile t visible to all waves

        f32x8 acc = {0.f, 0.f, 0.f, 0.f, 0.f, 0.f, 0.f, 0.f};
        const unsigned short* tile = cbT[buf];
        #pragma unroll
        for (int kc = 0; kc < 8; ++kc) {
            Frag Af;
            #pragma unroll
            for (int v = 0; v < 8; ++v)
                Af.u[v] = *(const unsigned*)(tile + (moff + ln) * LP + kc * 32 + kmap(v, kg));
            acc = __builtin_amdgcn_wmma_f32_16x16x32_bf16(false, Af.v, false, Bf[kc].v,
                                                          (short)0, acc, false, false);
        }
        #pragma unroll
        for (int v = 0; v < 8; ++v) {
            int mrow = moff + v + 8 * kg;
            float s = nrm[buf][mrow] - 2.0f * acc[v];
            int g = t * TROWS + mrow;
            if (s < bestv || (s == bestv && g < besti)) { bestv = s; besti = g; }
        }
        __syncthreads();                        // compute done before buf is overwritten
    }

    // lane l holds column N=l&15: combine the two lane-halves
    float ov = swap16_f(bestv); int oi = swap16_i(besti);
    if (ov < bestv || (ov == bestv && oi < besti)) { bestv = ov; besti = oi; }

    if (lane < 16) { rv[mgrp][nsub][lane] = bestv; ri[mgrp][nsub][lane] = besti; }
    __syncthreads();
    if (mgrp == 0 && lane < 16) {
        float a = rv[0][nsub][lane], b = rv[1][nsub][lane];
        int ia = ri[0][nsub][lane], ib = ri[1][nsub][lane];
        int win = (b < a || (b == a && ib < ia)) ? ib : ia;
        idx_out[nbase + lane] = win;
    }
}

// ---------------- gather + rotation trick + loss partials ----------------
__global__ void k_rotate(const float* __restrict__ zf, const float* __restrict__ cb,
                         const int* __restrict__ idx, float* __restrict__ out,
                         float* __restrict__ lpart, float* __restrict__ idxf) {
    __shared__ float szz[256], sqq[256], szq[256];
    int i = blockIdx.x, c = threadIdx.x;
    int id = idx[i];
    float zc = zf[(size_t)i * CDIM + c];
    float qc = cb[(size_t)id * CDIM + c];
    szz[c] = zc * zc; sqq[c] = qc * qc; szq[c] = zc * qc;
    __syncthreads();
    for (int o = 128; o > 0; o >>= 1) {
        if (c < o) { szz[c] += szz[c + o]; sqq[c] += sqq[c + o]; szq[c] += szq[c + o]; }
        __syncthreads();
    }
    float zz = szz[0], qq = sqq[0], zq = szq[0];
    float nsrc = sqrtf(zz), ntgt = sqrtf(qq);
    float nsrc_c = fmaxf(nsrc, 1e-6f), ntgt_c = fmaxf(ntgt, 1e-6f);
    float su = zz / (nsrc_c * nsrc_c), sq = qq / (ntgt_c * ntgt_c);
    float udq = zq / (nsrc_c * ntgt_c);
    float wn_c = fmaxf(sqrtf(su + sq + 2.f * udq), 1e-6f);
    float ew = (zz / nsrc_c + zq / ntgt_c) / wn_c;   // e.w (w normalized)
    float eu = zz / nsrc_c;                          // e.u
    float uc = zc / nsrc_c, qn = qc / ntgt_c;
    float rot = zc - 2.f * ew * ((uc + qn) / wn_c) + 2.f * eu * qn;
    rot *= ntgt / nsrc_c;

    int b = i >> 10, hw = i & 1023;
    out[((size_t)(b * CDIM + c) << 10) + hw] = rot;
    if (c == 0) {
        lpart[i] = zz + qq - 2.f * zq;   // sum (z - zq)^2 for this row
        idxf[i] = (float)id;
    }
}

__global__ void k_loss(const float* __restrict__ lpart, float* __restrict__ lossout) {
    __shared__ float s[256];
    int c = threadIdx.x;
    float a = 0.f;
    for (int j = 0; j < 64; ++j) a += lpart[c * 64 + j];
    s[c] = a;
    __syncthreads();
    for (int o = 128; o > 0; o >>= 1) { if (c < o) s[c] += s[c + o]; __syncthreads(); }
    if (c == 0) lossout[0] = 1.25f * s[0] / (float)(NZ * CDIM);   // (1+BETA)*mean
}

extern "C" void kernel_launch(void* const* d_in, const int* in_sizes, int n_in,
                              void* d_out, int out_size, void* d_ws, size_t ws_size,
                              hipStream_t stream) {
    const float* z = (const float*)d_in[0];
    const float* F = (const float*)d_in[1];
    const float* W = (const float*)d_in[2];
    float* out = (float*)d_out;

    char* ws = (char*)d_ws;
    size_t o = 0;
    float*          zf  = (float*)(ws + o);          o += (size_t)NZ * CDIM * 4;
    unsigned short* zb  = (unsigned short*)(ws + o); o += (size_t)NZ * CDIM * 2;
    unsigned short* Fb  = (unsigned short*)(ws + o); o += (size_t)N_E * EDIM * 2;
    unsigned short* Wb  = (unsigned short*)(ws + o); o += (size_t)CDIM * EDIM * 2;
    float*          cb  = (float*)(ws + o);          o += (size_t)N_E * CDIM * 4;
    unsigned short* cbb = (unsigned short*)(ws + o); o += (size_t)N_E * CDIM * 2;
    float*          cbn = (float*)(ws + o);          o += (size_t)N_E * 4;
    int*            idx = (int*)(ws + o);            o += (size_t)NZ * 4;
    float*          lp  = (float*)(ws + o);          o += (size_t)NZ * 4;
    (void)ws_size; (void)in_sizes; (void)n_in; (void)out_size;

    k_prep_z  <<<NZ, 256, 0, stream>>>(z, zf, zb);
    k_cvt_bf16<<<(N_E * EDIM) / 256, 256, 0, stream>>>(F, Fb, N_E * EDIM);
    k_cvt_bf16<<<(CDIM * EDIM) / 256, 256, 0, stream>>>(W, Wb, CDIM * EDIM);
    k_transform<<<N_E / 128, 256, 0, stream>>>(Fb, Wb, cb);
    k_cbnorm  <<<N_E, 256, 0, stream>>>(cb, cbb, cbn);
    k_dist    <<<NZ / 64, 256, 0, stream>>>(zb, cbb, cbn, idx);
    k_rotate  <<<NZ, 256, 0, stream>>>(zf, cb, idx, out, lp, out + (size_t)NZ * CDIM + 1);
    k_loss    <<<1, 256, 0, stream>>>(lp, out + (size_t)NZ * CDIM);
}
